// GeneratorBlock_27367531610759
// MI455X (gfx1250) — compile-verified, matrix operand order checked by reading, not run on hardware
//
#include <hip/hip_runtime.h>

typedef __attribute__((ext_vector_type(16))) _Float16 v16h;
typedef __attribute__((ext_vector_type(8)))  float    v8f;

#define BATCH 8
#define CIN   512
#define COUT  512
#define HH_   64
#define WW_   64
#define HWPX  4096        // 64*64
#define KTOT  4608        // 512*9
#define TM    128
#define TN    128
#define TK    64          // K per staged tile (2 WMMA k-steps)
#define ASTR  68          // padded K stride (halves) for LDS tiles
#define TILEH (TM * ASTR) // halves per A (or B) tile = 8704
#define LRELU 0.2f

// ---------------------------------------------------------------------------
// 1) EqualizedLinear styles: A[b][j] = sum_i w[b][i] * sW[j][i] * c + sb[j]
// ---------------------------------------------------------------------------
__global__ void eq_linear_k(const float* __restrict__ w,
                            const float* __restrict__ sW,
                            const float* __restrict__ sb,
                            float* __restrict__ A) {
  int t = blockIdx.x * blockDim.x + threadIdx.x;   // 0 .. BATCH*512-1
  int b = t >> 9;
  int j = t & 511;
  const float c = 0.044194173824159216f;           // 1/sqrt(512)
  const float* wr = w + b * 512;
  const float* sr = sW + j * 512;
  float acc = 0.f;
  #pragma unroll 8
  for (int i = 0; i < 512; ++i) acc += wr[i] * sr[i];
  A[t] = acc * c + sb[j];
}

// ---------------------------------------------------------------------------
// 2) Weight modulate + demodulate -> f16 GEMM-A  [b][cout][cin*9]
// ---------------------------------------------------------------------------
__global__ __launch_bounds__(256) void modw_k(const float* __restrict__ convW,
                                              const float* __restrict__ A,
                                              _Float16* __restrict__ out) {
  int blk = blockIdx.x;
  int b  = blk >> 9;
  int co = blk & 511;
  const float cmod = 0.014731391274719739f;        // 1/sqrt(4608)
  const float* wrow = convW + (size_t)co * KTOT;
  const float* arow = A + b * CIN;

  float vals[18];
  float ss = 0.f;
  #pragma unroll
  for (int i = 0; i < 18; ++i) {
    int k = threadIdx.x + i * 256;                 // 0..4607
    float wv = wrow[k] * cmod * arow[k / 9];
    vals[i] = wv;
    ss += wv * wv;
  }
  __shared__ float red[256];
  red[threadIdx.x] = ss;
  __syncthreads();
  for (int s = 128; s > 0; s >>= 1) {
    if (threadIdx.x < s) red[threadIdx.x] += red[threadIdx.x + s];
    __syncthreads();
  }
  float sinv = rsqrtf(red[0] + 1e-8f);
  _Float16* orow = out + ((size_t)b * COUT + co) * KTOT;
  #pragma unroll
  for (int i = 0; i < 18; ++i) {
    int k = threadIdx.x + i * 256;
    orow[k] = (_Float16)(vals[i] * sinv);
  }
}

// ---------------------------------------------------------------------------
// 3) f32 -> f16 cast (4 elems/thread, n % 4 == 0)
// ---------------------------------------------------------------------------
__global__ void cvt_k(const float* __restrict__ in, _Float16* __restrict__ out, int n4) {
  int i = blockIdx.x * blockDim.x + threadIdx.x;
  if (i >= n4) return;
  const float4 v = ((const float4*)in)[i];
  _Float16* o = out + i * 4;
  o[0] = (_Float16)v.x; o[1] = (_Float16)v.y;
  o[2] = (_Float16)v.z; o[3] = (_Float16)v.w;
}

// ---------------------------------------------------------------------------
// 4) Implicit-im2col WMMA GEMM + noise/bias/leaky-relu epilogue
//    grid = (N/128, M/128, B); 256 threads = 8 waves
//    wave -> 32(M) x 64(N) region = 2x4 tiles of 16x16
//    TK=64 double-buffered LDS; incremental (cin,off) im2col indexing
// ---------------------------------------------------------------------------
__global__ __launch_bounds__(256) void gemm_conv3x3_k(
    const _Float16* __restrict__ Wmod,   // [B][COUT][KTOT]
    const _Float16* __restrict__ Xh,     // [B][CIN][64][64]
    const float* __restrict__ noise,     // [B][1][64][64]
    const float* __restrict__ scale_noise,
    const float* __restrict__ bias,      // [COUT]
    float* __restrict__ Yf,              // [B][COUT][HWPX] or nullptr
    _Float16* __restrict__ Yh) {         // [B][COUT][HWPX]
  extern __shared__ _Float16 smem[];     // [2 bufs][A tile | B tile], 4*8704 halves

  const int b  = blockIdx.z;
  const int m0 = blockIdx.y * TM;
  const int n0 = blockIdx.x * TN;

  const int tid  = threadIdx.x;
  const int wave = tid >> 5;
  const int lane = tid & 31;
  const int wm = wave >> 1;              // 0..3 -> m offset wm*32
  const int wn = wave & 1;               // 0..1 -> n offset wn*64
  const int l16 = lane & 15;
  const int khalf = (lane >> 4) << 4;    // 0 or 16 (frag K half per lane group)

  v8f acc[2][4];
  #pragma unroll
  for (int mt = 0; mt < 2; ++mt)
    #pragma unroll
    for (int nt = 0; nt < 4; ++nt) acc[mt][nt] = (v8f){0.f,0.f,0.f,0.f,0.f,0.f,0.f,0.f};

  // ---- staging maps ----
  const int arow = tid >> 1;             // A: 0..127
  const int akh  = (tid & 1) << 4;       // A: 0 or 16 (segments add +0/+32)
  const int kb   = tid >> 3;             // B: base k-row 0..31 (units: kb, kb+32)
  const int nseg = (tid & 7) << 4;       // B: 0,16,...,112
  const int hbase = (n0 + nseg) >> 6;    // loop-invariant pixel row
  const int w0    = (n0 + nseg) & 63;    // loop-invariant pixel col (mult of 16)

  // loop-carried (cin, off) per B staging unit: k = k0 + kb + 32*j
  int cinS[2], offS[2];
  cinS[0] = kb / 9;        offS[0] = kb - cinS[0] * 9;
  { int k1 = kb + 32; cinS[1] = k1 / 9; offS[1] = k1 - cinS[1] * 9; }

  auto stage = [&](int buf, int k0) {
    _Float16* dA = smem + buf * (2 * TILEH);
    _Float16* dB = dA + TILEH;
    // A: 128 x 64 weight tile, contiguous f16 rows
    const _Float16* ga = Wmod + ((size_t)b * COUT + (m0 + arow)) * KTOT + k0 + akh;
    #pragma unroll
    for (int seg = 0; seg < 2; ++seg)
      #pragma unroll
      for (int i = 0; i < 16; ++i)
        dA[arow * ASTR + akh + seg * 32 + i] = ga[seg * 32 + i];
    if ((tid & 63) == 0 && k0 + TK < KTOT)
      __builtin_prefetch(ga + TK, 0, 3);          // near-scope global_prefetch_b8
    // B: implicit im2col, stored transposed [n][k]
    #pragma unroll
    for (int j = 0; j < 2; ++j) {
      const int cin  = cinS[j];
      const int off  = offS[j];
      const int krow = kb + 32 * j;
      const int o3   = off / 3;                   // 0..2 (tiny const-range div)
      const int dy   = o3 - 1;
      const int dx   = off - o3 * 3 - 1;
      const int hh   = hbase + dy;
      const bool rowok = ((unsigned)hh < (unsigned)HH_);
      const _Float16* src = Xh + ((size_t)b * CIN + cin) * HWPX + (hh << 6);
      if (rowok && (unsigned)(w0 + dx) <= (unsigned)(WW_ - 16)) {
        #pragma unroll
        for (int i = 0; i < 16; ++i)
          dB[(nseg + i) * ASTR + krow] = src[w0 + dx + i];
      } else {
        #pragma unroll
        for (int i = 0; i < 16; ++i) {
          int wx = w0 + i + dx;
          _Float16 v = (_Float16)0.f;
          if (rowok && (unsigned)wx < (unsigned)WW_) v = src[wx];
          dB[(nseg + i) * ASTR + krow] = v;
        }
      }
    }
  };
  auto advance = [&]() {                          // k += TK(=64) = 7*9 + 1
    #pragma unroll
    for (int j = 0; j < 2; ++j) {
      cinS[j] += 7; offS[j] += 1;
      if (offS[j] >= 9) { offS[j] -= 9; ++cinS[j]; }
    }
  };
  auto compute = [&](int buf) {
    const _Float16* cA = smem + buf * (2 * TILEH);
    const _Float16* cB = cA + TILEH;
    #pragma unroll
    for (int kk = 0; kk < TK; kk += 32) {
      union { v16h v; _Float16 h[16]; } af[2], bf[4];
      #pragma unroll
      for (int mt = 0; mt < 2; ++mt) {
        const int row = wm * 32 + mt * 16 + l16;
        #pragma unroll
        for (int i = 0; i < 16; ++i) af[mt].h[i] = cA[row * ASTR + kk + khalf + i];
      }
      #pragma unroll
      for (int nt = 0; nt < 4; ++nt) {
        const int col = wn * 64 + nt * 16 + l16;
        #pragma unroll
        for (int i = 0; i < 16; ++i) bf[nt].h[i] = cB[col * ASTR + kk + khalf + i];
      }
      #pragma unroll
      for (int mt = 0; mt < 2; ++mt)
        #pragma unroll
        for (int nt = 0; nt < 4; ++nt)
          acc[mt][nt] = __builtin_amdgcn_wmma_f32_16x16x32_f16(
              false, af[mt].v, false, bf[nt].v, (short)0, acc[mt][nt], false, false);
    }
  };

  // ---- software-pipelined main loop ----
  stage(0, 0);
  advance();
  __syncthreads();
  const int NIT = KTOT / TK;                      // 72
  for (int it = 0; it < NIT; ++it) {
    if (it + 1 < NIT) { stage((it + 1) & 1, (it + 1) * TK); advance(); }
    compute(it & 1);
    __syncthreads();
  }

  // ---- epilogue: + scale*noise + bias, leaky relu, write f32 + f16 ----
  const float sn = scale_noise[0];
  const int mlane = (lane >> 4) << 3;             // lanes 16-31 -> M += 8
  #pragma unroll
  for (int mt = 0; mt < 2; ++mt) {
    #pragma unroll
    for (int nt = 0; nt < 4; ++nt) {
      union { v8f v; float f[8]; } cu;
      cu.v = acc[mt][nt];
      const int n = n0 + wn * 64 + nt * 16 + l16;
      const float nz = sn * noise[b * HWPX + n];
      #pragma unroll
      for (int r = 0; r < 8; ++r) {
        const int m = m0 + wm * 32 + mt * 16 + r + mlane;
        float v = cu.f[r] + nz + bias[m];
        v = (v > 0.f) ? v : LRELU * v;
        const size_t oi = ((size_t)b * COUT + m) * HWPX + n;
        if (Yf) Yf[oi] = v;
        Yh[oi] = (_Float16)v;
      }
    }
  }
}

// ---------------------------------------------------------------------------
// 5) RGB: modulated 1x1 conv (no demod) + leaky relu
// ---------------------------------------------------------------------------
__global__ __launch_bounds__(256) void rgb_k(const float* __restrict__ A3,
                                             const float* __restrict__ rgbW,   // [3][512]
                                             const float* __restrict__ rgb_bias,
                                             const _Float16* __restrict__ Xh2, // [B][512][HWPX]
                                             float* __restrict__ out) {        // [B][3][HWPX]
  const int b = blockIdx.y;
  const float cm = 0.044194173824159216f;          // 1/sqrt(512*1*1)
  __shared__ _Float16 wsh[3 * 512];
  #pragma unroll
  for (int i = 0; i < 6; ++i) {
    int idx = threadIdx.x + i * 256;
    int r = idx >> 9, c = idx & 511;
    wsh[idx] = (_Float16)(rgbW[r * 512 + c] * cm * A3[b * 512 + c]);
  }
  __syncthreads();

  const int n = blockIdx.x * 256 + threadIdx.x;
  const _Float16* xp = Xh2 + (size_t)b * 512 * HWPX + n;
  float s0 = 0.f, s1 = 0.f, s2 = 0.f;
  #pragma unroll 4
  for (int c = 0; c < 512; ++c) {
    float xv = (float)xp[(size_t)c * HWPX];
    s0 += xv * (float)wsh[c];
    s1 += xv * (float)wsh[512 + c];
    s2 += xv * (float)wsh[1024 + c];
  }
  float* op = out + (size_t)b * 3 * HWPX + n;
  float v0 = s0 + rgb_bias[0]; op[0]        = (v0 > 0.f) ? v0 : LRELU * v0;
  float v1 = s1 + rgb_bias[1]; op[HWPX]     = (v1 > 0.f) ? v1 : LRELU * v1;
  float v2 = s2 + rgb_bias[2]; op[2 * HWPX] = (v2 > 0.f) ? v2 : LRELU * v2;
}

// ---------------------------------------------------------------------------
extern "C" void kernel_launch(void* const* d_in, const int* in_sizes, int n_in,
                              void* d_out, int out_size, void* d_ws, size_t ws_size,
                              hipStream_t stream) {
  const float* x           = (const float*)d_in[0];
  const float* w           = (const float*)d_in[1];
  const float* noise       = (const float*)d_in[2];   // [2][8][1][64][64]
  const float* s1_style_W  = (const float*)d_in[3];
  const float* s1_style_b  = (const float*)d_in[4];
  const float* s1_conv_W   = (const float*)d_in[5];
  const float* s1_sn       = (const float*)d_in[6];
  const float* s1_bias     = (const float*)d_in[7];
  const float* s2_style_W  = (const float*)d_in[8];
  const float* s2_style_b  = (const float*)d_in[9];
  const float* s2_conv_W   = (const float*)d_in[10];
  const float* s2_sn       = (const float*)d_in[11];
  const float* s2_bias     = (const float*)d_in[12];
  const float* rgb_style_W = (const float*)d_in[13];
  const float* rgb_style_b = (const float*)d_in[14];
  const float* rgb_conv_W  = (const float*)d_in[15];
  const float* rgb_bias    = (const float*)d_in[16];

  float* out_x   = (float*)d_out;                         // [8][512][4096]
  float* out_rgb = out_x + (size_t)BATCH * COUT * HWPX;   // [8][3][4096]

  // workspace layout
  char* ws = (char*)d_ws;
  float*    A1   = (float*)(ws);                          // 8*512 f32 each
  float*    A2   = A1 + BATCH * 512;
  float*    A3   = A2 + BATCH * 512;
  _Float16* Wmod = (_Float16*)(ws + 65536);                         // 37.75 MB (reused)
  _Float16* bufX = (_Float16*)(ws + 65536 + (size_t)37748736);      // 33.55 MB
  _Float16* bufY = (_Float16*)(ws + 65536 + (size_t)37748736 + (size_t)33554432);

  const size_t smem_bytes = (size_t)4 * TILEH * sizeof(_Float16);   // 69632 B

  // 1) styles
  eq_linear_k<<<dim3(16), 256, 0, stream>>>(w, s1_style_W,  s1_style_b,  A1);
  eq_linear_k<<<dim3(16), 256, 0, stream>>>(w, s2_style_W,  s2_style_b,  A2);
  eq_linear_k<<<dim3(16), 256, 0, stream>>>(w, rgb_style_W, rgb_style_b, A3);

  // 2) cast x -> f16
  const int n4 = BATCH * CIN * HWPX / 4;
  cvt_k<<<dim3((n4 + 255) / 256), 256, 0, stream>>>(x, bufX, n4);

  // 3) style block 1
  modw_k<<<dim3(BATCH * COUT), 256, 0, stream>>>(s1_conv_W, A1, Wmod);
  gemm_conv3x3_k<<<dim3(HWPX / TN, COUT / TM, BATCH), 256, smem_bytes, stream>>>(
      Wmod, bufX, noise, s1_sn, s1_bias, nullptr, bufY);

  // 4) style block 2 (f32 -> d_out, f16 -> bufX for RGB)
  modw_k<<<dim3(BATCH * COUT), 256, 0, stream>>>(s2_conv_W, A2, Wmod);
  gemm_conv3x3_k<<<dim3(HWPX / TN, COUT / TM, BATCH), 256, smem_bytes, stream>>>(
      Wmod, bufY, noise + (size_t)BATCH * HWPX, s2_sn, s2_bias, out_x, bufX);

  // 5) RGB
  rgb_k<<<dim3(HWPX / 256, BATCH), 256, 0, stream>>>(A3, rgb_conv_W, rgb_bias, bufX, out_rgb);
}
